// PonderAttention_48292612276455
// MI455X (gfx1250) — compile-verified, matrix operand order checked by reading, not run on hardware
//
#include <hip/hip_runtime.h>
#include <hip/hip_bf16.h>
#include <math.h>

// Problem constants (reference: B=2, T=2048, C=2048, H=16, HD=128)
#define B_  2
#define T_  2048
#define C_  2048
#define H_  16
#define HD_ 128
#define M_  (B_*T_)          // 4096 rows for projection GEMMs
#define T16 (T_/16)          // 128 q-tiles per (b,h)

typedef __bf16 bf16;
typedef __attribute__((ext_vector_type(16))) __bf16 v16bf;
typedef __attribute__((ext_vector_type(8)))  __bf16 v8bf;
typedef __attribute__((ext_vector_type(8)))  float  v8f;
typedef __attribute__((ext_vector_type(4)))  unsigned int u32x4;
typedef __attribute__((ext_vector_type(8)))  int i32x8;
typedef __attribute__((ext_vector_type(4)))  int i32x4;

#if defined(__has_builtin)
#if __has_builtin(__builtin_amdgcn_tensor_load_to_lds) && \
    __has_builtin(__builtin_amdgcn_s_wait_tensorcnt)
#define HAVE_TDM 1
#endif
#endif
#ifndef HAVE_TDM
#define HAVE_TDM 0
#endif

union FragBF { v16bf v; v8bf h[2]; bf16 s[16]; };

// Load a 16-bit A/B WMMA fragment for one lane.
// Per ISA 16-bit 16x32 layout: lane holds row (ln); elements 0..7 -> K = hi*8+0..7,
// elements 8..15 -> K = 16+hi*8+0..7  (two contiguous 16-byte chunks).
__device__ __forceinline__ v16bf load_frag(const bf16* __restrict__ rowbase, int hi) {
  FragBF f;
  f.h[0] = *reinterpret_cast<const v8bf*>(rowbase + hi * 8);
  f.h[1] = *reinterpret_cast<const v8bf*>(rowbase + 16 + hi * 8);
  return f.v;
}

__device__ __forceinline__ v8f wmma_bf16(v16bf a, v16bf b, v8f c) {
  return __builtin_amdgcn_wmma_f32_16x16x32_bf16(false, a, false, b, (short)0, c,
                                                 false, false);
}

#if HAVE_TDM
// Issue a TDM 2D tile load (global -> LDS). D# built per ISA cdna5 8.3/8.4:
// group0: count=1, lds_addr, global_addr[56:0], type=2 ("image").
// group1: data_size=2B, tensor_dim0/1, tile_dim0/1, tensor_dim0_stride.
// One instruction per wave; EXEC ignored; completion tracked by TENSORcnt.
__device__ __forceinline__ void tdm_load_2d(unsigned lds_off,
                                            const bf16* __restrict__ gptr,
                                            unsigned tile_w, unsigned tile_h,
                                            unsigned row_stride_elems) {
  unsigned long long ga = (unsigned long long)(uintptr_t)gptr;
  u32x4 g0;
  g0.x = 1u;                                      // count=1 (valid user descriptor)
  g0.y = lds_off;                                 // lds_addr (bytes)
  g0.z = (unsigned)(ga & 0xFFFFFFFFu);            // global_addr[31:0]
  g0.w = (unsigned)((ga >> 32) & 0x01FFFFFFu)     // global_addr[56:32]
         | (2u << 30);                            // type=2
  unsigned td0 = row_stride_elems;                // tensor_dim0 >= tile_w (no OOB)
  unsigned td1 = 0x00100000u;                     // large tensor_dim1 (no OOB)
  unsigned long long s0 = (unsigned long long)row_stride_elems;  // dim0 stride
  i32x8 g1;
  g1[0] = (int)(1u << 16);                        // data_size=1 (2 bytes)
  g1[1] = (int)((td0 & 0xFFFFu) << 16);           // [63:48] = tensor_dim0.lo
  g1[2] = (int)((td0 >> 16) | ((td1 & 0xFFFFu) << 16));
  g1[3] = (int)((td1 >> 16) | (tile_w << 16));    // [127:112] = tile_dim0
  g1[4] = (int)(tile_h & 0xFFFFu);                // tile_dim1; tile_dim2=0 (2D)
  g1[5] = (int)(s0 & 0xFFFFFFFFu);
  g1[6] = (int)((s0 >> 32) & 0xFFFFu);            // stride1=0
  g1[7] = 0;
  i32x4 gz = {0, 0, 0, 0};
#if defined(__clang_major__) && __clang_major__ >= 23
  i32x8 gz8 = {0, 0, 0, 0, 0, 0, 0, 0};
  __builtin_amdgcn_tensor_load_to_lds(g0, g1, gz, gz, gz8, 0);
#else
  __builtin_amdgcn_tensor_load_to_lds(g0, g1, gz, gz, 0);
#endif
}
#endif  // HAVE_TDM

// ---------------------------------------------------------------- convert
__global__ void f32_to_bf16_kernel(const float* __restrict__ in,
                                   bf16* __restrict__ out, size_t n) {
  size_t i = (size_t)blockIdx.x * blockDim.x + threadIdx.x;
  size_t stride = (size_t)gridDim.x * blockDim.x;
  for (; i < n; i += stride) out[i] = (bf16)in[i];
}

// ---------------------------------------------------------------- GEMM: Y = A * W^T
// A: (Mdim x Kdim) bf16 row-major, W: (Ndim x Kdim) bf16 row-major.
// Block tile 128x128, 8 waves; wave tile 32(m) x 64(n) = 8 WMMAs per K-step.
// A/W K-slabs (128x32) staged by the Tensor Data Mover, double-buffered so the
// DMA of slab k+32 overlaps the 64 block-WMMAs on slab k.
template <bool OUT_F32>
__global__ __launch_bounds__(256)
void gemm_awt_kernel(const bf16* __restrict__ A, const bf16* __restrict__ W,
                     void* __restrict__ Y, int Mdim, int Ndim, int Kdim) {
  __shared__ __align__(16) bf16 lds_ab[2][2][128][32];  // [buf][A/W][row][k]

  const int tid = threadIdx.x;
  const int lane = tid & 31;
  const int ln = lane & 15;
  const int hi = lane >> 4;
  const int wid = tid >> 5;
  const int wm = wid & 3;        // 4 waves along m
  const int wn = wid >> 2;       // 2 waves along n
  const int mblk = blockIdx.x * 128;
  const int nblk = blockIdx.y * 128;

  v8f z = {};
  v8f acc[2][4];
  #pragma unroll
  for (int s = 0; s < 2; ++s)
    #pragma unroll
    for (int t = 0; t < 4; ++t) acc[s][t] = z;

  const bf16* Abase = A + (size_t)mblk * Kdim;
  const bf16* Wbase = W + (size_t)nblk * Kdim;

#if HAVE_TDM
  // LDS byte offsets of the four slabs (single shared var => block base 0).
  const unsigned SLAB = 128u * 32u * 2u;          // 8 KB
  if (wid == 0) {
    tdm_load_2d(0u * SLAB, Abase, 32u, 128u, (unsigned)Kdim);
    tdm_load_2d(1u * SLAB, Wbase, 32u, 128u, (unsigned)Kdim);
    __builtin_amdgcn_s_wait_tensorcnt(0);
  }
  __syncthreads();

  int cur = 0;
  for (int k = 0; k < Kdim; k += 32) {
    const int nxt = cur ^ 1;
    // Prefetch next K-slab via TDM while this slab is being consumed.
    if (wid == 0 && (k + 32) < Kdim) {
      tdm_load_2d((unsigned)(nxt * 2) * SLAB,     Abase + k + 32, 32u, 128u,
                  (unsigned)Kdim);
      tdm_load_2d((unsigned)(nxt * 2 + 1) * SLAB, Wbase + k + 32, 32u, 128u,
                  (unsigned)Kdim);
    }

    v16bf af[2], bfr[4];
    #pragma unroll
    for (int s = 0; s < 2; ++s)
      af[s] = load_frag(&lds_ab[cur][0][wm * 32 + s * 16 + ln][0], hi);
    #pragma unroll
    for (int t = 0; t < 4; ++t)
      bfr[t] = load_frag(&lds_ab[cur][1][wn * 64 + t * 16 + ln][0], hi);

    #pragma unroll
    for (int s = 0; s < 2; ++s)
      #pragma unroll
      for (int t = 0; t < 4; ++t)
        acc[s][t] = wmma_bf16(af[s], bfr[t], acc[s][t]);

    if (wid == 0) __builtin_amdgcn_s_wait_tensorcnt(0);
    __syncthreads();     // next slab resident + all reads of cur done
    cur = nxt;
  }
#else
  for (int k = 0; k < Kdim; k += 32) {
    __syncthreads();
    #pragma unroll
    for (int i = 0; i < 2; ++i) {
      int c = tid * 2 + i;           // 512 chunks of 8 bf16 per matrix
      int row = c >> 2;
      int co = (c & 3) * 8;
      *reinterpret_cast<v8bf*>(&lds_ab[0][0][row][co]) =
          *reinterpret_cast<const v8bf*>(Abase + (size_t)row * Kdim + k + co);
      *reinterpret_cast<v8bf*>(&lds_ab[0][1][row][co]) =
          *reinterpret_cast<const v8bf*>(Wbase + (size_t)row * Kdim + k + co);
    }
    __syncthreads();

    v16bf af[2], bfr[4];
    #pragma unroll
    for (int s = 0; s < 2; ++s)
      af[s] = load_frag(&lds_ab[0][0][wm * 32 + s * 16 + ln][0], hi);
    #pragma unroll
    for (int t = 0; t < 4; ++t)
      bfr[t] = load_frag(&lds_ab[0][1][wn * 64 + t * 16 + ln][0], hi);
    #pragma unroll
    for (int s = 0; s < 2; ++s)
      #pragma unroll
      for (int t = 0; t < 4; ++t)
        acc[s][t] = wmma_bf16(af[s], bfr[t], acc[s][t]);
  }
#endif

  // D layout: VGPR r, lanes 0-15 -> row r, lanes 16-31 -> row r+8; col = ln.
  #pragma unroll
  for (int s = 0; s < 2; ++s)
    #pragma unroll
    for (int t = 0; t < 4; ++t)
      #pragma unroll
      for (int r = 0; r < 8; ++r) {
        size_t row = (size_t)(mblk + wm * 32 + s * 16 + r + 8 * hi);
        size_t col = (size_t)(nblk + wn * 64 + t * 16 + ln);
        float val = acc[s][t][r];
        if (OUT_F32)
          reinterpret_cast<float*>(Y)[row * Ndim + col] = val;
        else
          reinterpret_cast<bf16*>(Y)[row * Ndim + col] = (bf16)val;
      }
}

// ---------------------------------------------------------------- RoPE (in place)
// X layout: (B, T, H, HD) == (B, T, C). One thread per rotation pair.
__global__ void rope_kernel(bf16* __restrict__ X, float scale) {
  const size_t total = (size_t)B_ * T_ * H_ * (HD_ / 2);
  size_t i = (size_t)blockIdx.x * blockDim.x + threadIdx.x;
  if (i >= total) return;
  int d = (int)(i % (HD_ / 2)); size_t r = i / (HD_ / 2);
  int h = (int)(r % H_); r /= H_;
  int t = (int)(r % T_); size_t b = r / T_;
  size_t base = ((b * T_ + t) * (size_t)C_) + (size_t)h * HD_;
  float u1 = (float)X[base + d];
  float u2 = (float)X[base + d + HD_ / 2];
  float invf = __expf(-((float)d / 64.0f) * 9.210340372f);  // 10000^(-d/64)
  float ang = (float)t * invf;
  float c = __cosf(ang), s = __sinf(ang);
  X[base + d]           = (bf16)((u1 * c - u2 * s) * scale);
  X[base + d + HD_ / 2] = (bf16)((u2 * c + u1 * s) * scale);
}

// ---------------------------------------------------------------- flash attention
// One wave owns 16 query rows of one (b,h). Online softmax, KV chunks of 32.
__global__ __launch_bounds__(256)
void flash_attn_kernel(const bf16* __restrict__ Q, const bf16* __restrict__ K,
                       const bf16* __restrict__ V, bf16* __restrict__ O) {
  __shared__ __align__(16) bf16 pbuf[8][16][32];   // per-wave P staging (D->A relayout)

  const int lane = threadIdx.x & 31;
  const int ln = lane & 15;
  const int hi = lane >> 4;
  const int wid = threadIdx.x >> 5;

  int tile = blockIdx.x * 8 + wid;                 // 4096 tiles total
  int b = tile / (H_ * T16);
  int rem = tile % (H_ * T16);
  int h = rem / T16;
  int q0 = (rem % T16) * 16;

  const size_t headoff = (size_t)h * HD_;
  const bf16* Qb = Q + (size_t)b * T_ * C_ + headoff;
  const bf16* Kb = K + (size_t)b * T_ * C_ + headoff;
  const bf16* Vb = V + (size_t)b * T_ * C_ + headoff;

  // Q resident: 16 rows x HD=128 -> 4 A-fragments (K-steps of 32)
  v16bf qf[4];
  const bf16* qrow = Qb + (size_t)(q0 + ln) * C_;
  #pragma unroll
  for (int kk = 0; kk < 4; ++kk) qf[kk] = load_frag(qrow + kk * 32, hi);

  v8f z = {};
  v8f o[8];
  #pragma unroll
  for (int j = 0; j < 8; ++j) o[j] = z;
  float mrow[8], lrow[8];
  #pragma unroll
  for (int r = 0; r < 8; ++r) { mrow[r] = -3.0e38f; lrow[r] = 0.0f; }

  const int kend = q0 + 16;
  for (int k0 = 0; k0 < kend; k0 += 32) {
    // Warm L2/L0 for the next KV chunk while this one is computed.
    __builtin_prefetch(Kb + (size_t)(k0 + 32 + ln) * C_, 0, 0);
    __builtin_prefetch(Vb + (size_t)(k0 + 32 + ln) * C_, 0, 0);

    // ---- S = Q * K^T for this 16x32 chunk (two 16x16 n-tiles).
    v8f s[2]; s[0] = z; s[1] = z;
    #pragma unroll
    for (int tn = 0; tn < 2; ++tn) {
      const bf16* krow = Kb + (size_t)(k0 + tn * 16 + ln) * C_;
      v16bf kf[4];
      #pragma unroll
      for (int kk = 0; kk < 4; ++kk) kf[kk] = load_frag(krow + kk * 32, hi);
      #pragma unroll
      for (int kk = 0; kk < 4; ++kk) s[tn] = wmma_bf16(qf[kk], kf[kk], s[tn]);
    }

    // ---- causal mask + chunk row-max
    float cm[8];
    #pragma unroll
    for (int r = 0; r < 8; ++r) cm[r] = -3.0e38f;
    #pragma unroll
    for (int tn = 0; tn < 2; ++tn)
      #pragma unroll
      for (int r = 0; r < 8; ++r) {
        int colkv = k0 + tn * 16 + ln;
        int rowq = q0 + r + 8 * hi;
        float sv = s[tn][r];
        if (colkv > rowq) sv = -3.0e38f;
        s[tn][r] = sv;
        cm[r] = fmaxf(cm[r], sv);
      }
    #pragma unroll
    for (int off = 1; off < 16; off <<= 1)
      #pragma unroll
      for (int r = 0; r < 8; ++r)
        cm[r] = fmaxf(cm[r], __shfl_xor(cm[r], off, 32));

    // ---- online softmax update
    float alpha[8];
    #pragma unroll
    for (int r = 0; r < 8; ++r) {
      float mn = fmaxf(mrow[r], cm[r]);
      alpha[r] = __expf(mrow[r] - mn);
      mrow[r] = mn;
    }
    float rs[8];
    #pragma unroll
    for (int r = 0; r < 8; ++r) rs[r] = 0.0f;
    #pragma unroll
    for (int tn = 0; tn < 2; ++tn)
      #pragma unroll
      for (int r = 0; r < 8; ++r) {
        float p = __expf(s[tn][r] - mrow[r]);
        rs[r] += p;
        pbuf[wid][r + 8 * hi][tn * 16 + ln] = (bf16)p;   // D-layout -> [m][k] in LDS
      }
    #pragma unroll
    for (int off = 1; off < 16; off <<= 1)
      #pragma unroll
      for (int r = 0; r < 8; ++r) rs[r] += __shfl_xor(rs[r], off, 32);
    #pragma unroll
    for (int r = 0; r < 8; ++r) lrow[r] = lrow[r] * alpha[r] + rs[r];
    #pragma unroll
    for (int j = 0; j < 8; ++j)
      #pragma unroll
      for (int r = 0; r < 8; ++r) o[j][r] *= alpha[r];

    // LDS RAW within wave: DS ops are in-order, but be explicit before b128 reads
    asm volatile("s_wait_dscnt 0" ::: "memory");

    // ---- read P back as an A-fragment (16x32)
    FragBF pf;
    pf.h[0] = *reinterpret_cast<const v8bf*>(&pbuf[wid][ln][hi * 8]);
    pf.h[1] = *reinterpret_cast<const v8bf*>(&pbuf[wid][ln][16 + hi * 8]);

    // ---- O += P * V  (8 n-tiles across HD=128)
    #pragma unroll
    for (int j = 0; j < 8; ++j) {
      FragBF bv;
      #pragma unroll
      for (int e = 0; e < 16; ++e) {
        int kv = ((e < 8) ? e : e + 8) + hi * 8;   // B-frag K mapping
        bv.s[e] = Vb[(size_t)(k0 + kv) * C_ + j * 16 + ln];
      }
      o[j] = wmma_bf16(pf.v, bv.v, o[j]);
    }
  }

  // ---- normalize and write out in (B,T,C) layout
  #pragma unroll
  for (int r = 0; r < 8; ++r) {
    float inv = 1.0f / lrow[r];
    size_t row = (size_t)b * T_ + q0 + r + 8 * hi;
    #pragma unroll
    for (int j = 0; j < 8; ++j)
      O[row * C_ + headoff + j * 16 + ln] = (bf16)(o[j][r] * inv);
  }
}

// ---------------------------------------------------------------- launch
extern "C" void kernel_launch(void* const* d_in, const int* in_sizes, int n_in,
                              void* d_out, int out_size, void* d_ws, size_t ws_size,
                              hipStream_t stream) {
  (void)in_sizes; (void)n_in; (void)out_size; (void)ws_size;
  const float* x  = (const float*)d_in[0];
  const float* Wq = (const float*)d_in[1];
  const float* Wk = (const float*)d_in[2];
  const float* Wv = (const float*)d_in[3];
  const float* Wo = (const float*)d_in[4];
  // d_in[5] (mask) unused: causality is computed analytically.

  char* ws = (char*)d_ws;
  const size_t xbytes = (size_t)M_ * C_ * sizeof(bf16);   // 16 MB
  const size_t wbytes = (size_t)C_ * C_ * sizeof(bf16);   //  8 MB
  bf16* xb  = (bf16*)ws;            ws += xbytes;
  bf16* Wqb = (bf16*)ws;            ws += wbytes;
  bf16* Wkb = (bf16*)ws;            ws += wbytes;
  bf16* Wvb = (bf16*)ws;            ws += wbytes;
  bf16* Wob = (bf16*)ws;            ws += wbytes;
  bf16* Qb  = (bf16*)ws;            ws += xbytes;
  bf16* Kb  = (bf16*)ws;            ws += xbytes;
  bf16* Vb  = (bf16*)ws;            ws += xbytes;
  bf16* Ab  = (bf16*)ws;            ws += xbytes;   // ~112 MB total

  f32_to_bf16_kernel<<<2048, 256, 0, stream>>>(x,  xb,  (size_t)M_ * C_);
  f32_to_bf16_kernel<<<2048, 256, 0, stream>>>(Wq, Wqb, (size_t)C_ * C_);
  f32_to_bf16_kernel<<<2048, 256, 0, stream>>>(Wk, Wkb, (size_t)C_ * C_);
  f32_to_bf16_kernel<<<2048, 256, 0, stream>>>(Wv, Wvb, (size_t)C_ * C_);
  f32_to_bf16_kernel<<<2048, 256, 0, stream>>>(Wo, Wob, (size_t)C_ * C_);

  dim3 gg(M_ / 128, C_ / 128);
  gemm_awt_kernel<false><<<gg, 256, 0, stream>>>(xb, Wqb, (void*)Qb, M_, C_, C_);
  gemm_awt_kernel<false><<<gg, 256, 0, stream>>>(xb, Wkb, (void*)Kb, M_, C_, C_);
  gemm_awt_kernel<false><<<gg, 256, 0, stream>>>(xb, Wvb, (void*)Vb, M_, C_, C_);

  const size_t pairs = (size_t)B_ * T_ * H_ * (HD_ / 2);
  const float qscale = 0.08838834764831845f;   // 1/sqrt(HD=128), folded into Q
  rope_kernel<<<(unsigned)((pairs + 255) / 256), 256, 0, stream>>>(Qb, qscale);
  rope_kernel<<<(unsigned)((pairs + 255) / 256), 256, 0, stream>>>(Kb, 1.0f);

  flash_attn_kernel<<<(B_ * H_ * T16) / 8, 256, 0, stream>>>(Qb, Kb, Vb, Ab);

  gemm_awt_kernel<true><<<gg, 256, 0, stream>>>(Ab, Wob, d_out, M_, C_, C_);
}
